// Scale_Attention_3994319585902
// MI455X (gfx1250) — matrix operand order repaired and checked by reading, not compiled, not run on hardware
//
#include <hip/hip_runtime.h>
#include <hip/hip_bf16.h>

// ---- problem constants (B=1) ----
#define TT   4
#define CC   128
#define NN   4096
#define HDD  64
#define LN_EPS 1e-5f
#define KSTEP 32
#define KROWSTRIDE 72   // padded LDS row stride for K tile (32x64 f16)
#define VROWSTRIDE 40   // padded LDS row stride for V tile (64x32 f16)
#define PROWSTRIDE 40   // padded LDS row stride for P bounce (16x32 f16)

typedef __attribute__((ext_vector_type(16))) _Float16 v16h;
typedef __attribute__((ext_vector_type(8)))  _Float16 v8h;
typedef __attribute__((ext_vector_type(8)))  float    v8f;

__device__ __forceinline__ v8f zero8() {
  v8f z = {0.f,0.f,0.f,0.f,0.f,0.f,0.f,0.f};
  return z;
}

// A-operand (16x32 f16) load, contiguous K order matching B:
// lane half h: elems [k0+16h .. k0+16h+15]
__device__ __forceinline__ v16h wmma_a_load(const _Float16* __restrict__ row, int k0, int h) {
  union { v16h v; v8h p[2]; } u;
  u.p[0] = *(const v8h*)(row + k0 + 16*h);
  u.p[1] = *(const v8h*)(row + k0 + 16*h + 8);
  return u.v;
}

// B-operand (32x16 f16): per lane, 16 contiguous K values (K = 16h + 0..15), 32B aligned
__device__ __forceinline__ v16h wmma_b_load(const _Float16* __restrict__ p) {
  return *(const v16h*)p;
}

// same but only 16B-aligned (padded LDS rows): two explicit b128 loads
__device__ __forceinline__ v16h wmma_b_load16(const _Float16* __restrict__ p) {
  union { v16h v; v8h q[2]; } u;
  u.q[0] = *(const v8h*)p;
  u.q[1] = *(const v8h*)(p + 8);
  return u.v;
}

__device__ __forceinline__ v8f wmma32(v16h a, v16h b, v8f c) {
  return __builtin_amdgcn_wmma_f32_16x16x32_f16(false, a, false, b, (short)0, c, false, false);
}

// store D tile transposed: lane writes 8 contiguous f16 over rows m0+8h..+7
__device__ __forceinline__ void store_tr8(_Float16* dst, v8f acc) {
  union { v8h v; _Float16 e[8]; } u;
  #pragma unroll
  for (int r = 0; r < 8; ++r) u.e[r] = (_Float16)acc[r];
  *(v8h*)dst = u.v;
}

// CDNA5 async copy: 16B global -> LDS, tracked by ASYNCcnt.
// Low 32 bits of a flat shared pointer are the LDS byte address (ISA 10.2 aperture rules).
__device__ __forceinline__ void async_ld16(_Float16* lds_ptr, const _Float16* gptr) {
  unsigned loff = (unsigned)(uintptr_t)lds_ptr;
  unsigned long long ga = (unsigned long long)(uintptr_t)gptr;
  asm volatile("global_load_async_to_lds_b128 %0, %1, off"
               :: "v"(loff), "v"(ga) : "memory");
}

// ---------------- kernel 1: f32 weights -> f16 ----------------
__global__ void prep_w_kernel(const float* __restrict__ qw, const float* __restrict__ kvw,
                              _Float16* __restrict__ wq, _Float16* __restrict__ wkv) {
  for (int i = threadIdx.x; i < CC*CC; i += blockDim.x) {
    wq[i]  = (_Float16)qw[i];
    wkv[i] = (_Float16)kvw[i];
  }
}

// ---------------- kernel 2: LayerNorm (+GELU) with transpose to (N,C) f16 ----------------
__global__ void ln_gelu_kernel(const float* __restrict__ x0, const float* __restrict__ x1,
                               const float* __restrict__ x2,
                               const float* __restrict__ g, const float* __restrict__ bb,
                               _Float16* __restrict__ f0, _Float16* __restrict__ f1,
                               _Float16* __restrict__ f2) {
  const int n = blockIdx.x * blockDim.x + threadIdx.x;
  const int t = blockIdx.y;
  const int s = blockIdx.z;
  const float* x = (s == 0) ? x0 : (s == 1) ? x1 : x2;
  _Float16*  d  = (s == 0) ? f0 : (s == 1) ? f1 : f2;
  const float* xp = x + (size_t)t * CC * NN + n;

  float sum = 0.f, sq = 0.f;
  #pragma unroll 4
  for (int c = 0; c < CC; ++c) {
    float v = xp[(size_t)c * NN];
    sum += v; sq += v * v;
  }
  const float mean = sum * (1.f / CC);
  const float var  = sq * (1.f / CC) - mean * mean;
  const float rstd = rsqrtf(var + LN_EPS);

  _Float16* dp = d + (size_t)t * NN * CC + (size_t)n * CC;
  #pragma unroll 1          // keep one erff expansion set, not 16 (I-cache)
  for (int c8 = 0; c8 < CC / 8; ++c8) {
    union { v8h v; _Float16 e[8]; } u;
    #pragma unroll
    for (int k = 0; k < 8; ++k) {
      const int c = c8 * 8 + k;
      float v = (xp[(size_t)c * NN] - mean) * rstd * g[c] + bb[c];
      if (s != 0) v = 0.5f * v * (1.f + erff(v * 0.70710678118f));  // exact GELU
      u.e[k] = (_Float16)v;
    }
    *(v8h*)(dp + c8 * 8) = u.v;   // 16B store per 8 channels
  }
}

// ---------------- kernel 3: projections (one wave = 16 rows x 128 cols, K=128) ----------------
__global__ void proj_kernel(const _Float16* __restrict__ f0, const _Float16* __restrict__ f1,
                            const _Float16* __restrict__ f2,
                            const _Float16* __restrict__ wq, const _Float16* __restrict__ wkv,
                            const float* __restrict__ qb, const float* __restrict__ kvb,
                            _Float16* __restrict__ qh,  _Float16* __restrict__ p1T,
                            _Float16* __restrict__ p2h, _Float16* __restrict__ p2T) {
  const int lane = threadIdx.x & 31, l16 = lane & 15, h = lane >> 4;
  const int rt = blockIdx.x, t = blockIdx.y, s = blockIdx.z;
  const int base = rt * 16;

  const _Float16* in = ((s == 0) ? f0 : (s == 1) ? f1 : f2) + (size_t)t * NN * CC;
  const _Float16* w  = (s == 0) ? wq : wkv;
  const float* bias  = (s == 0) ? qb : kvb;

  const _Float16* arow = in + (size_t)(base + l16) * CC;
  v16h a[4];
  #pragma unroll
  for (int kc = 0; kc < 4; ++kc) a[kc] = wmma_a_load(arow, kc * 32, h);

  #pragma unroll
  for (int ct = 0; ct < 8; ++ct) {
    v8f c = zero8();
    #pragma unroll
    for (int kc = 0; kc < 4; ++kc) {
      v16h b = wmma_b_load(w + (size_t)(ct * 16 + l16) * CC + kc * 32 + 16 * h);
      c = wmma32(a[kc], b, c);
    }
    const float bv = bias[ct * 16 + l16];
    #pragma unroll
    for (int r = 0; r < 8; ++r) c[r] += bv;

    if (s == 0) {
      _Float16* o = qh + (size_t)t * NN * CC;
      #pragma unroll
      for (int r = 0; r < 8; ++r)
        o[(size_t)(base + r + 8 * h) * CC + ct * 16 + l16] = (_Float16)c[r];
    } else if (s == 1) {
      _Float16* o = p1T + (size_t)t * CC * NN;
      store_tr8(o + (size_t)(ct * 16 + l16) * NN + base + 8 * h, c);
    } else {
      _Float16* o = p2h + (size_t)t * NN * CC;
      #pragma unroll
      for (int r = 0; r < 8; ++r)
        o[(size_t)(base + r + 8 * h) * CC + ct * 16 + l16] = (_Float16)c[r];
      _Float16* oT = p2T + (size_t)t * CC * NN;
      store_tr8(oT + (size_t)(ct * 16 + l16) * NN + base + 8 * h, c);
    }
  }
}

// ---------------- kernel 4: ctx = k1^T @ v1 (64x64, K=4096) ----------------
__global__ void ctx_kernel(const _Float16* __restrict__ p1T, _Float16* __restrict__ ctxT) {
  const int lane = threadIdx.x & 31, l16 = lane & 15, h = lane >> 4;
  const int w = threadIdx.x >> 5;
  const int t = blockIdx.x;
  const _Float16* pt   = p1T + (size_t)t * CC * NN;
  const _Float16* arow = pt + (size_t)(w * 16 + l16) * NN;  // k1T rows 0..63

  v8f acc[4] = {zero8(), zero8(), zero8(), zero8()};
  #pragma unroll 2
  for (int nb = 0; nb < NN; nb += 32) {
    v16h a = wmma_a_load(arow, nb, h);
    #pragma unroll
    for (int et = 0; et < 4; ++et) {
      v16h b = wmma_b_load(pt + (size_t)(64 + et * 16 + l16) * NN + nb + 16 * h);
      acc[et] = wmma32(a, b, acc[et]);
    }
  }
  _Float16* ct = ctxT + (size_t)t * HDD * HDD;
  #pragma unroll
  for (int et = 0; et < 4; ++et)
    store_tr8(ct + (size_t)(et * 16 + l16) * HDD + w * 16 + 8 * h, acc[et]);
}

// stage one 32-key K block (32x64, row-major, padded) and V block (64x32 from v2T, padded)
// into LDS via async copies: 2 x 16B per thread (256 threads) = 4KB + 4KB payload.
__device__ __forceinline__ void stage_kv(const _Float16* __restrict__ kmat,
                                         const _Float16* __restrict__ vT,
                                         _Float16* kb_dst, _Float16* vb_dst,
                                         int kb, int tid) {
  const int krow = tid >> 3, kch = tid & 7;   // 32 rows x 8 chunks
  async_ld16(kb_dst + krow * KROWSTRIDE + kch * 8,
             kmat + (size_t)(kb + krow) * CC + kch * 8);
  const int vrow = tid >> 2, vch = tid & 3;   // 64 rows x 4 chunks
  async_ld16(vb_dst + vrow * VROWSTRIDE + vch * 8,
             vT + (size_t)vrow * NN + kb + vch * 8);
}

// ---------------- kernel 5: o1 = q1@ctx*scale + flash softmax o2 ----------------
// 8 waves, 128 query rows per block; K/V double-buffered through LDS via async copies.
__global__ void flash_kernel(const _Float16* __restrict__ qh, const _Float16* __restrict__ p2h,
                             const _Float16* __restrict__ p2T, const _Float16* __restrict__ ctxT,
                             float* __restrict__ out) {
  __shared__ __align__(16) _Float16 kbuf[2 * 32 * KROWSTRIDE];
  __shared__ __align__(16) _Float16 vbuf[2 * 64 * VROWSTRIDE];
  __shared__ __align__(16) _Float16 pbuf[8 * 16 * PROWSTRIDE];
  const int tid  = threadIdx.x;
  const int lane = tid & 31, l16 = lane & 15, h = lane >> 4;
  const int w = tid >> 5;
  const int t = blockIdx.y;
  const int nb0 = blockIdx.x * 128 + w * 16;
  const float scale = 0.125f;  // hd^-0.5

  const _Float16* qrow = qh + (size_t)t * NN * CC + (size_t)(nb0 + l16) * CC;
  v16h aq1[2], aq2[2];
  aq1[0] = wmma_a_load(qrow, 0,  h);  aq1[1] = wmma_a_load(qrow, 32, h);
  aq2[0] = wmma_a_load(qrow, 64, h);  aq2[1] = wmma_a_load(qrow, 96, h);

  const _Float16* kmat = p2h + (size_t)t * NN * CC;                    // k2 = cols 0..63
  const _Float16* vT   = p2T + (size_t)t * CC * NN + (size_t)HDD * NN; // v2T rows 64..127

  // prologue: fill both buffers
  stage_kv(kmat, vT, kbuf, vbuf, 0, tid);
  stage_kv(kmat, vT, kbuf + 32 * KROWSTRIDE, vbuf + 64 * VROWSTRIDE, KSTEP, tid);

  // o1 = q1 @ ctx (K=64) while the first tiles are in flight
  const _Float16* ct = ctxT + (size_t)t * HDD * HDD;
  v8f o1[4] = {zero8(), zero8(), zero8(), zero8()};
  #pragma unroll
  for (int et = 0; et < 4; ++et)
    #pragma unroll
    for (int kc = 0; kc < 2; ++kc) {
      v16h b = wmma_b_load(ct + (size_t)(et * 16 + l16) * HDD + kc * 32 + 16 * h);
      o1[et] = wmma32(aq1[kc], b, o1[et]);
    }

  float m_r[8], l_r[8];
  #pragma unroll
  for (int r = 0; r < 8; ++r) { m_r[r] = -1e30f; l_r[r] = 0.f; }
  v8f oacc[4] = {zero8(), zero8(), zero8(), zero8()};
  _Float16* pb = pbuf + w * (16 * PROWSTRIDE);

  const int NITER = NN / KSTEP;
  #pragma unroll 1          // never unroll the 128-deep pipeline loop
  for (int j = 0; j < NITER; ++j) {
    _Float16* kb_c = kbuf + (j & 1) * 32 * KROWSTRIDE;
    _Float16* vb_c = vbuf + (j & 1) * 64 * VROWSTRIDE;
    // retire this buffer's 2 async ops (prefetched buffer may stay in flight)
    if (j + 1 < NITER) asm volatile("s_wait_asynccnt 2" ::: "memory");
    else               asm volatile("s_wait_asynccnt 0" ::: "memory");
    __syncthreads();

    v8f s0 = zero8(), s1 = zero8();
    #pragma unroll
    for (int kc = 0; kc < 2; ++kc) {
      v16h b0 = wmma_b_load16(kb_c + (     l16) * KROWSTRIDE + kc * 32 + 16 * h);
      v16h b1 = wmma_b_load16(kb_c + (16 + l16) * KROWSTRIDE + kc * 32 + 16 * h);
      s0 = wmma32(aq2[kc], b0, s0);
      s1 = wmma32(aq2[kc], b1, s1);
    }
    float alpha[8];
    #pragma unroll
    for (int r = 0; r < 8; ++r) {
      float a0 = s0[r] * scale, a1 = s1[r] * scale;
      float rm = fmaxf(a0, a1);
      rm = fmaxf(rm, __shfl_xor(rm, 1, 32));
      rm = fmaxf(rm, __shfl_xor(rm, 2, 32));
      rm = fmaxf(rm, __shfl_xor(rm, 4, 32));
      rm = fmaxf(rm, __shfl_xor(rm, 8, 32));
      const float mn = fmaxf(m_r[r], rm);
      const float al = __expf(m_r[r] - mn);
      const float p0 = __expf(a0 - mn);
      const float p1 = __expf(a1 - mn);
      float rs = p0 + p1;
      rs += __shfl_xor(rs, 1, 32);
      rs += __shfl_xor(rs, 2, 32);
      rs += __shfl_xor(rs, 4, 32);
      rs += __shfl_xor(rs, 8, 32);
      l_r[r] = l_r[r] * al + rs;
      m_r[r] = mn;
      alpha[r] = al;
      pb[(r + 8 * h) * PROWSTRIDE + l16]      = (_Float16)p0;
      pb[(r + 8 * h) * PROWSTRIDE + 16 + l16] = (_Float16)p1;
    }
    #pragma unroll
    for (int dt = 0; dt < 4; ++dt)
      #pragma unroll
      for (int r = 0; r < 8; ++r) oacc[dt][r] *= alpha[r];

    asm volatile("s_wait_dscnt 0" ::: "memory");  // intra-wave LDS RAW fence
    v16h ap = wmma_a_load(pb + l16 * PROWSTRIDE, 0, h);
    #pragma unroll
    for (int dt = 0; dt < 4; ++dt) {
      v16h bv = wmma_b_load16(vb_c + (dt * 16 + l16) * VROWSTRIDE + 16 * h);
      oacc[dt] = wmma32(ap, bv, oacc[dt]);
    }

    __syncthreads();  // everyone done reading buf (j&1) before it is overwritten
    if (j + 2 < NITER)
      stage_kv(kmat, vT, kb_c, vb_c, (j + 2) * KSTEP, tid);
  }

  // reference reshape is a raw reinterpret of the (N,C) buffer -> write row-major
  float* ob = out + (size_t)t * CC * NN;
  #pragma unroll
  for (int dt = 0; dt < 4; ++dt)
    #pragma unroll
    for (int r = 0; r < 8; ++r) {
      const size_t n = (size_t)(nb0 + r + 8 * h);
      ob[n * CC + dt * 16 + l16]       = o1[dt][r] * scale;
      ob[n * CC + 64 + dt * 16 + l16]  = oacc[dt][r] / l_r[r];
    }
}

extern "C" void kernel_launch(void* const* d_in, const int* in_sizes, int n_in,
                              void* d_out, int out_size, void* d_ws, size_t ws_size,
                              hipStream_t stream) {
  (void)in_sizes; (void)n_in; (void)out_size; (void)ws_size;
  const float* x0  = (const float*)d_in[0];
  const float* x1  = (const float*)d_in[1];
  const float* x2  = (const float*)d_in[2];
  const float* g   = (const float*)d_in[3];
  const float* bb  = (const float*)d_in[4];
  const float* qw  = (const float*)d_in[5];
  const float* qb  = (const float*)d_in[6];
  const float* kvw = (const float*)d_in[7];
  const float* kvb = (const float*)d_in[8];
  float* out = (float*)d_out;

  const size_t NC = (size_t)NN * CC;
  _Float16* wq   = (_Float16*)d_ws;        // C*C
  _Float16* wkv  = wq   + (size_t)CC * CC; // C*C
  _Float16* f0   = wkv  + (size_t)CC * CC; // T*N*C each below
  _Float16* f1   = f0   + (size_t)TT * NC;
  _Float16* f2   = f1   + (size_t)TT * NC;
  _Float16* qh   = f2   + (size_t)TT * NC;
  _Float16* p1T  = qh   + (size_t)TT * NC;
  _Float16* p2h  = p1T  + (size_t)TT * NC;
  _Float16* p2T  = p2h  + (size_t)TT * NC;
  _Float16* ctxT = p2T  + (size_t)TT * NC; // T*64*64
  // total ws use: ~29.5 MB of f16

  prep_w_kernel<<<dim3(1), 256, 0, stream>>>(qw, kvw, wq, wkv);
  ln_gelu_kernel<<<dim3(NN / 256, TT, 3), 256, 0, stream>>>(x0, x1, x2, g, bb, f0, f1, f2);
  proj_kernel<<<dim3(NN / 16, TT, 3), 32, 0, stream>>>(f0, f1, f2, wq, wkv, qb, kvb,
                                                       qh, p1T, p2h, p2T);
  ctx_kernel<<<dim3(TT), 128, 0, stream>>>(p1T, ctxT);
  flash_kernel<<<dim3(NN / 128, TT), 256, 0, stream>>>(qh, p2h, p2T, ctxT, out);
}
